// SpikingNeuralSubstrate_12987981103590
// MI455X (gfx1250) — matrix operand order repaired and checked by reading, not compiled, not run on hardware
//
#include <hip/hip_runtime.h>

typedef __attribute__((ext_vector_type(16))) __bf16 v16bf;
typedef __attribute__((ext_vector_type(8)))  __bf16 bf16x8;
typedef __attribute__((ext_vector_type(8)))  float  v8f;
typedef __attribute__((ext_vector_type(4)))  float  f32x4;
typedef int i32x4 __attribute__((vector_size(16)));   // matches async builtin param

#define T_STEPS 16
#define BATCH   512
#define N_IN    2048
#define N_HID   8192
#define N_OUT   2048
#define DECAY   0.9f
#define V_TH    1.0f

// Block tile: 64 (batch) x 128 (neurons), K-step 32.
#define BM 64
#define BN 128
#define BK 32
#define LDSTR 40   // padded LDS row stride in elements: 80 B = 20 banks

// ---------------------------------------------------------------------------
// CDNA5 async global->LDS copies (ASYNCcnt-tracked DMA), with sync fallback.
// Signature (from hipcc diagnostic): first param is int4* in addrspace(1).
// ---------------------------------------------------------------------------
#if defined(__has_builtin)
#if __has_builtin(__builtin_amdgcn_global_load_async_to_lds_b128)
#define USE_ASYNC_LDS 1
#endif
#endif
#ifndef USE_ASYNC_LDS
#define USE_ASYNC_LDS 0
#endif

__device__ __forceinline__ void async_cp_b128(__bf16* dst_lds, const __bf16* src_g) {
#if USE_ASYNC_LDS
    __builtin_amdgcn_global_load_async_to_lds_b128(
        (__attribute__((address_space(1))) i32x4*)src_g,
        (__attribute__((address_space(3))) i32x4*)dst_lds,
        0, 0);
#else
    *(bf16x8*)dst_lds = *(const bf16x8*)src_g;
#endif
}

__device__ __forceinline__ void async_wait_all() {
#if USE_ASYNC_LDS
#if __has_builtin(__builtin_amdgcn_s_wait_asynccnt)
    __builtin_amdgcn_s_wait_asynccnt(0);
#else
    asm volatile("s_wait_asynccnt 0x0" ::: "memory");
#endif
#endif
}

// ---------------------------------------------------------------------------
// Split fp32 weights into hi/lo bf16 planes: w ~= hi + lo (≈16 mantissa bits).
// ---------------------------------------------------------------------------
__global__ void split_bf16_kernel(const float* __restrict__ W,
                                  __bf16* __restrict__ hi,
                                  __bf16* __restrict__ lo,
                                  size_t n) {
    size_t stride = (size_t)gridDim.x * blockDim.x;
    for (size_t i = (size_t)blockIdx.x * blockDim.x + threadIdx.x; i < n; i += stride) {
        float w  = W[i];
        __bf16 h = (__bf16)w;
        hi[i] = h;
        lo[i] = (__bf16)(w - (float)h);
    }
}

// ---------------------------------------------------------------------------
// Fused GEMM (spikes @ W.T via WMMA bf16 hi/lo) + LIF integrate/fire/reset.
// Double-buffered LDS: async copies for k-step t+1 overlap WMMAs on k-step t.
// ---------------------------------------------------------------------------
template<bool FROM_X>
__global__ __launch_bounds__(256)
void lif_gemm_kernel(const void* __restrict__ Asrc,
                     const __bf16* __restrict__ Whi,
                     const __bf16* __restrict__ Wlo,
                     float*  __restrict__ v,
                     __bf16* __restrict__ s_bf,   // spike out (hidden layer), or null
                     float*  __restrict__ s_f32,  // spike out (output layer), or null
                     int K, int Nn, int first) {
    constexpr int ATILE = BM * LDSTR;          // 2560 elems (5 KB)
    constexpr int BTILE = BN * LDSTR;          // 5120 elems (10 KB)
    constexpr int BUF   = ATILE + 2 * BTILE;   // A + Whi + Wlo tile
    __shared__ __bf16 smem[2 * BUF];           // double buffered (~50 KB)

    const int tid  = threadIdx.x;
    const int lane = tid & 31;
    const int wave = tid >> 5;
    const int wm   = wave >> 2;   // 0..1  (M direction)
    const int wn   = wave & 3;    // 0..3  (N direction)

    const int n_blk = blockIdx.x * BN;
    const int m_blk = blockIdx.y * BM;

    v8f acc[2][2] = {};

    // Cooperative staging coordinates.
    const int a_row = tid >> 2;          // A: 64 rows x 32 k, 8 elems/thread
    const int a_col = (tid & 3) * 8;
    const int b_row = tid >> 1;          // B: 128 rows x 32 k, 16 elems/thread/plane
    const int b_col = (tid & 1) * 16;

    // A-fragment layout (16x32 bf16): lanes 0-15 hold k 0-15,
    // lanes 16-31 hold k 16-31 (contiguous), for row m = lane&15.
    const int frag_row0 = wm * 32 + (lane & 15);
    const int a_koff    = (lane & 16) ? 16 : 0;
    // B-fragment layout (32x16 bf16): lane = column n (lane&15),
    // lanes 0-15 hold k 0-15, lanes 16-31 hold k 16-31 (contiguous per lane).
    const int b_koff = (lane & 16) ? 16 : 0;
    const int b_n    = lane & 15;

    // ---- stage one k-tile (A spikes + both weight planes) into smem[buf] ----
    auto stage = [&](int buf, int kt) {
        __bf16* A  = smem + buf * BUF;
        __bf16* Bh = A + ATILE;
        __bf16* Bl = Bh + BTILE;

        // Weight planes: pure byte moves -> async DMA into LDS.
        {
            const size_t g = (size_t)(n_blk + b_row) * K + kt + b_col;
            __bf16* dh = &Bh[b_row * LDSTR + b_col];
            __bf16* dl = &Bl[b_row * LDSTR + b_col];
            async_cp_b128(dh,     Whi + g);
            async_cp_b128(dh + 8, Whi + g + 8);
            async_cp_b128(dl,     Wlo + g);
            async_cp_b128(dl + 8, Wlo + g + 8);
        }
        // A operand.
        if (FROM_X) {
            // fp32 drive -> threshold -> bf16 {0,1} (needs VALU, manual path)
            const float* xsrc = (const float*)Asrc;
            const float* p = xsrc + (size_t)(m_blk + a_row) * K + kt + a_col;
            f32x4 x0 = *(const f32x4*)p;
            f32x4 x1 = *(const f32x4*)(p + 4);
            bf16x8 av;
#pragma unroll
            for (int e = 0; e < 4; ++e) av[e]     = (__bf16)((x0[e] > 0.f) ? 1.f : 0.f);
#pragma unroll
            for (int e = 0; e < 4; ++e) av[4 + e] = (__bf16)((x1[e] > 0.f) ? 1.f : 0.f);
            *(bf16x8*)(&A[a_row * LDSTR + a_col]) = av;
        } else {
            // spikes already bf16 {0,1} -> async DMA
            const __bf16* ssrc = (const __bf16*)Asrc;
            async_cp_b128(&A[a_row * LDSTR + a_col],
                          ssrc + (size_t)(m_blk + a_row) * K + kt + a_col);
        }
    };

    // Prologue: fill buffer 0, then pipeline.
    stage(0, 0);
    async_wait_all();
    __syncthreads();

    int cur = 0;
    for (int kt = 0; kt < K; kt += BK) {
        const int nxt = cur ^ 1;
        if (kt + BK < K) stage(nxt, kt + BK);   // DMA next tile during compute

        const __bf16* A  = smem + cur * BUF;
        const __bf16* Bh = A + ATILE;
        const __bf16* Bl = Bh + BTILE;

        // ---- A fragments from LDS ----
        v16bf afrag[2];
#pragma unroll
        for (int i = 0; i < 2; ++i) {
            const __bf16* base = &A[(frag_row0 + i * 16) * LDSTR + a_koff];
            bf16x8 c0 = *(const bf16x8*)base;
            bf16x8 c1 = *(const bf16x8*)(base + 8);
            afrag[i] = __builtin_shufflevector(c0, c1,
                0, 1, 2, 3, 4, 5, 6, 7, 8, 9, 10, 11, 12, 13, 14, 15);
        }

        // ---- B fragments (hi/lo) from LDS, 8 WMMAs ----
#pragma unroll
        for (int j = 0; j < 2; ++j) {
            const int rowb = (wn * 32 + j * 16 + b_n) * LDSTR + b_koff;
            bf16x8 h0 = *(const bf16x8*)(Bh + rowb);
            bf16x8 h1 = *(const bf16x8*)(Bh + rowb + 8);
            v16bf bh = __builtin_shufflevector(h0, h1,
                0, 1, 2, 3, 4, 5, 6, 7, 8, 9, 10, 11, 12, 13, 14, 15);
            bf16x8 l0 = *(const bf16x8*)(Bl + rowb);
            bf16x8 l1 = *(const bf16x8*)(Bl + rowb + 8);
            v16bf bl = __builtin_shufflevector(l0, l1,
                0, 1, 2, 3, 4, 5, 6, 7, 8, 9, 10, 11, 12, 13, 14, 15);
#pragma unroll
            for (int i = 0; i < 2; ++i) {
                acc[i][j] = __builtin_amdgcn_wmma_f32_16x16x32_bf16(
                    false, afrag[i], false, bh, (short)0, acc[i][j], false, false);
                acc[i][j] = __builtin_amdgcn_wmma_f32_16x16x32_bf16(
                    false, afrag[i], false, bl, (short)0, acc[i][j], false, false);
            }
        }

        async_wait_all();    // next tile landed in LDS
        __syncthreads();     // visible to all waves; cur tile reads done
        cur = nxt;
    }

    // ---- LIF epilogue: v' = v*decay + I; spike = (v' >= th); reset ----
    // C/D layout: lane n = lane&15 is the column; VGPR r holds row r (lanes
    // 0-15) or row 8+r (lanes 16-31).
    const int col_n   = lane & 15;
    const int row_off = (lane & 16) ? 8 : 0;
#pragma unroll
    for (int i = 0; i < 2; ++i) {
#pragma unroll
        for (int j = 0; j < 2; ++j) {
            const int n_g = n_blk + wn * 32 + j * 16 + col_n;
#pragma unroll
            for (int r = 0; r < 8; ++r) {
                const int m_g = m_blk + wm * 32 + i * 16 + row_off + r;
                const size_t idx = (size_t)m_g * Nn + n_g;
                float vold = first ? 0.f : v[idx];
                float vnew = vold * DECAY + acc[i][j][r];
                float sp   = (vnew >= V_TH) ? 1.f : 0.f;
                v[idx] = (sp > 0.f) ? 0.f : vnew;
                if (s_bf)  s_bf[idx]  = (__bf16)sp;
                if (s_f32) s_f32[idx] = sp;
            }
        }
    }
}

// ---------------------------------------------------------------------------
// Launch: convert weights once, then 16 steps x (hidden, output) fused kernels.
// Workspace: W1hi/lo 64MB + W2hi/lo 64MB + v_h 16MB + v_o 4MB + s_h 8MB ≈156MB.
// ---------------------------------------------------------------------------
extern "C" void kernel_launch(void* const* d_in, const int* in_sizes, int n_in,
                              void* d_out, int out_size, void* d_ws, size_t ws_size,
                              hipStream_t stream) {
    const float* x  = (const float*)d_in[0];   // [T, B, N_IN]
    const float* W1 = (const float*)d_in[1];   // [N_HID, N_IN]
    const float* W2 = (const float*)d_in[2];   // [N_OUT, N_HID]

    char* ws = (char*)d_ws;
    size_t off = 0;
    auto carve = [&](size_t bytes) -> void* {
        void* p = ws + off;
        off += (bytes + 255) & ~(size_t)255;
        return p;
    };
    __bf16* W1hi = (__bf16*)carve((size_t)N_HID * N_IN * sizeof(__bf16));
    __bf16* W1lo = (__bf16*)carve((size_t)N_HID * N_IN * sizeof(__bf16));
    __bf16* W2hi = (__bf16*)carve((size_t)N_OUT * N_HID * sizeof(__bf16));
    __bf16* W2lo = (__bf16*)carve((size_t)N_OUT * N_HID * sizeof(__bf16));
    float*  v_h  = (float*)carve((size_t)BATCH * N_HID * sizeof(float));
    float*  v_o  = (float*)carve((size_t)BATCH * N_OUT * sizeof(float));
    __bf16* s_h  = (__bf16*)carve((size_t)BATCH * N_HID * sizeof(__bf16));

    split_bf16_kernel<<<2048, 256, 0, stream>>>(W1, W1hi, W1lo, (size_t)N_HID * N_IN);
    split_bf16_kernel<<<2048, 256, 0, stream>>>(W2, W2hi, W2lo, (size_t)N_OUT * N_HID);

    dim3 block(256);
    dim3 g1(N_HID / BN, BATCH / BM);   // 64 x 8
    dim3 g2(N_OUT / BN, BATCH / BM);   // 16 x 8

    for (int t = 0; t < T_STEPS; ++t) {
        lif_gemm_kernel<true><<<g1, block, 0, stream>>>(
            (const void*)(x + (size_t)t * BATCH * N_IN),
            W1hi, W1lo, v_h, s_h, nullptr, N_IN, N_HID, t == 0);
        lif_gemm_kernel<false><<<g2, block, 0, stream>>>(
            (const void*)s_h,
            W2hi, W2lo, v_o, nullptr, (float*)d_out, N_HID, N_OUT, t == 0);
    }
}
